// MAB_21526376087658
// MI455X (gfx1250) — compile-verified, matrix operand order checked by reading, not run on hardware
//
#include <hip/hip_runtime.h>

typedef __bf16 bf16;
typedef __attribute__((ext_vector_type(16))) __bf16 v16bf;
typedef __attribute__((ext_vector_type(8)))  __bf16 v8bf;
typedef __attribute__((ext_vector_type(8)))  float  v8f;
typedef __attribute__((ext_vector_type(4)))  unsigned int u32x4;
typedef __attribute__((ext_vector_type(8)))  int i32x8;
typedef __attribute__((ext_vector_type(4)))  int i32x4;

#define B_SZ  16
#define NQ_   1024
#define NK_   1024
#define DV_   512
#define H_    4
#define DH_   128
#define MROWS (B_SZ * NQ_)   // 16384

// Tensor Data Mover available? (device pass only; host pass parses fallback)
#if defined(__AMDGCN__) && __has_builtin(__builtin_amdgcn_tensor_load_to_lds)
#define USE_TDM 1
#if __clang_major__ >= 23
#define TDM_LOAD(g0, g1, za, zb)                                         \
  do { i32x8 _z8 = {};                                                   \
       __builtin_amdgcn_tensor_load_to_lds((g0), (g1), (za), (zb), _z8, 0); } while (0)
#else
#define TDM_LOAD(g0, g1, za, zb)                                         \
  __builtin_amdgcn_tensor_load_to_lds((g0), (g1), (za), (zb), 0)
#endif
#else
#define USE_TDM 0
#endif

__device__ __forceinline__ bf16 f2bf(float f) { return (bf16)f; }   // v_cvt (RTNE)
__device__ __forceinline__ float bf2f(bf16 b) { return (float)b; }

__device__ __forceinline__ v8f wmma_bf16(v16bf a, v16bf b, v8f c) {
  // D(16x16,f32) = A(16x32,bf16) x B(32x16,bf16) + C
  return __builtin_amdgcn_wmma_f32_16x16x32_bf16(false, a, false, b,
                                                 (short)0, c, false, false);
}

// Per-lane A/B fragment per CDNA5 16-bit layout:
// lanes 0-15 hold K {0..7, 16..23}, lanes 16-31 hold K {8..15, 24..31}.
// base points at this lane's row (A: row m, B: column n); koff = (lane>>4)*8.
__device__ __forceinline__ v16bf load_frag(const bf16* base, int koff) {
  v8bf lo = *(const v8bf*)(base + koff);
  v8bf hi = *(const v8bf*)(base + koff + 16);
  v16bf r;
#pragma unroll
  for (int i = 0; i < 8; ++i) { r[i] = lo[i]; r[i + 8] = hi[i]; }
  return r;
}

// ---------------------------------------------------------------------------
// GEMM: Y(bf16)[M,512] = X(f32)[M,512] @ W(f32)[512,512]^T + bias
// block = 256 threads (8 waves), tile 128(M) x 64(N), K-step 32
// ---------------------------------------------------------------------------
__global__ __launch_bounds__(256, 2)
void proj_f32_kernel(const float* __restrict__ X, const float* __restrict__ W,
                     const float* __restrict__ bias, bf16* __restrict__ Y) {
  __shared__ __align__(16) bf16 As[128][32];
  __shared__ __align__(16) bf16 Bs[64][32];
  const int tid  = threadIdx.x;
  const int wave = tid >> 5, lane = tid & 31;
  const int half = lane >> 4, nc = lane & 15;
  const int m0 = blockIdx.x * 128, n0 = blockIdx.y * 64;
  const int koff = half * 8;
  v8f acc[4] = {};
  for (int k0 = 0; k0 < DV_; k0 += 32) {
#pragma unroll
    for (int i = 0; i < 4; ++i) {          // stage A: 128x32 f32 -> bf16
      int idx = tid + i * 256;             // 1024 float4 chunks
      int r = idx >> 3, c4 = idx & 7;
      float4 f = ((const float4*)(X + (size_t)(m0 + r) * DV_ + k0))[c4];
      bf16* dst = &As[r][c4 * 4];
      dst[0] = f2bf(f.x); dst[1] = f2bf(f.y); dst[2] = f2bf(f.z); dst[3] = f2bf(f.w);
    }
#pragma unroll
    for (int i = 0; i < 2; ++i) {          // stage B: 64x32 (rows of W = cols of W^T)
      int idx = tid + i * 256;             // 512 float4 chunks
      int r = idx >> 3, c4 = idx & 7;
      float4 f = ((const float4*)(W + (size_t)(n0 + r) * DV_ + k0))[c4];
      bf16* dst = &Bs[r][c4 * 4];
      dst[0] = f2bf(f.x); dst[1] = f2bf(f.y); dst[2] = f2bf(f.z); dst[3] = f2bf(f.w);
    }
    if (k0 + 32 < DV_)
      __builtin_prefetch(X + (size_t)(m0 + tid / 2) * DV_ + k0 + 32, 0, 1);
    __syncthreads();
    v16bf a = load_frag(&As[wave * 16 + nc][0], koff);
#pragma unroll
    for (int t = 0; t < 4; ++t)
      acc[t] = wmma_bf16(a, load_frag(&Bs[t * 16 + nc][0], koff), acc[t]);
    __syncthreads();
  }
#pragma unroll
  for (int t = 0; t < 4; ++t) {
    int col = n0 + t * 16 + nc;
    float bv = bias[col];
#pragma unroll
    for (int r = 0; r < 8; ++r) {
      int row = m0 + wave * 16 + half * 8 + r;
      Y[(size_t)row * DV_ + col] = f2bf(acc[t][r] + bv);
    }
  }
}

// ---------------------------------------------------------------------------
// Flash attention with log-weight bias and fused q-residual.
// grid = (NQ/64, H*B), block = 128 threads (4 waves), each wave: 16 q rows.
// K tile staged by the Tensor Data Mover (wave 0), V/weights staged manually.
// ---------------------------------------------------------------------------
__global__ __launch_bounds__(128, 2)
void attn_kernel(const bf16* __restrict__ qb, const bf16* __restrict__ kb,
                 const bf16* __restrict__ vb, const float* __restrict__ wts,
                 float* __restrict__ O) {
  __shared__ __align__(16) bf16  Ks[32][128];   // [key][dh]
  __shared__ __align__(16) bf16  Vt[128][32];   // [dh][key]  (transposed)
  __shared__ __align__(16) float Ws[64][32];    // [qrow][key] weights tile
  __shared__ __align__(16) bf16  Ps[4][16][32]; // per-wave P re-layout bounce
  const int tid  = threadIdx.x;
  const int wave = tid >> 5, lane = tid & 31;
  const int half = lane >> 4, nc = lane & 15;
  const int koff = half * 8;
  const int h = blockIdx.y >> 4, b = blockIdx.y & 15;   // hb = h*B + b
  const int q0blk = blockIdx.x * 64;
  const int qrow0 = q0blk + wave * 16;
  const float scale = 0.044194173824159216f;            // 1/sqrt(512)

  // preload q fragments (A operand), 16 rows x 128 dh = 4 K-chunks
  const size_t qlanebase = ((size_t)b * NQ_ + qrow0 + nc) * DV_ + h * DH_;
  v16bf aq[4];
#pragma unroll
  for (int c = 0; c < 4; ++c) aq[c] = load_frag(qb + qlanebase + c * 32, koff);

  v8f oacc[8] = {};
  float mrun[8], lrun[8];
#pragma unroll
  for (int r = 0; r < 8; ++r) { mrun[r] = -1e30f; lrun[r] = 0.f; }

  for (int k0 = 0; k0 < NK_; k0 += 32) {
#if USE_TDM
    if (wave == 0) {
      // D# for a 2D tile: 128 (dh, contiguous) x 32 (keys, stride 512 elem),
      // 2-byte elements, global -> LDS Ks.
      unsigned lds_off = (unsigned)(unsigned long long)&Ks[0][0]; // low 32 = LDS addr
      unsigned long long ga =
          (unsigned long long)(kb + ((size_t)b * NK_ + k0) * DV_ + h * DH_);
      u32x4 g0 = { 1u,                                    // count=1 (valid, user)
                   lds_off,                                // lds_addr
                   (unsigned)(ga & 0xFFFFFFFFu),           // global_addr[31:0]
                   (unsigned)((ga >> 32) & 0x1FFFFFFu) | 0x80000000u }; // hi + type=2
      i32x8 g1 = { (int)(1u << 16),                        // data_size = 2 bytes
                   (int)((unsigned)DH_ << 16),             // tensor_dim0 = 128 (lo16)
                   (int)((unsigned)NK_ << 16),             // dim0 hi=0, tensor_dim1 = 1024
                   (int)((unsigned)DH_ << 16),             // dim1 hi=0, tile_dim0 = 128
                   32,                                     // tile_dim1 = 32, tile_dim2 = 0
                   DV_,                                    // tensor_dim0_stride = 512 (lo32)
                   0, 0 };                                 // stride hi, dim1_stride unused
      i32x4 z4 = {};
      TDM_LOAD(g0, g1, z4, z4);
    }
#else
#pragma unroll
    for (int i = 0; i < 4; ++i) {          // fallback: manual K tile [32][128]
      int idx = tid + i * 128;
      int key = idx >> 4, c8 = idx & 15;
      v8bf v = *(const v8bf*)(kb + ((size_t)b * NK_ + k0 + key) * DV_ + h * DH_ + c8 * 8);
      *(v8bf*)(&Ks[key][c8 * 8]) = v;
    }
#endif
#pragma unroll
    for (int i = 0; i < 4; ++i) {          // stage V tile transposed [dh][key]
      int idx = tid + i * 128;
      int key = idx >> 4, c8 = idx & 15;
      v8bf v = *(const v8bf*)(vb + ((size_t)b * NK_ + k0 + key) * DV_ + h * DH_ + c8 * 8);
#pragma unroll
      for (int j = 0; j < 8; ++j) Vt[c8 * 8 + j][key] = v[j];
    }
#pragma unroll
    for (int i = 0; i < 4; ++i) {          // stage weights tile [64][32] f32
      int idx = tid + i * 128;             // 512 float4 chunks
      int r = idx >> 3, c4 = idx & 7;
      float4 f = ((const float4*)(wts + ((size_t)blockIdx.y * NQ_ + q0blk + r) * NK_ + k0))[c4];
      *(float4*)(&Ws[r][c4 * 4]) = f;
    }
#if USE_TDM
    if (wave == 0) __builtin_amdgcn_s_wait_tensorcnt(0);  // K tile landed in LDS
#endif
    __syncthreads();

    // scores: S(16x32) = q(16x128) @ K^T, 2 col tiles x 4 K-chunks
    v8f s[2] = {};
#pragma unroll
    for (int c = 0; c < 4; ++c) {
      s[0] = wmma_bf16(aq[c], load_frag(&Ks[nc][c * 32], koff), s[0]);
      s[1] = wmma_bf16(aq[c], load_frag(&Ks[16 + nc][c * 32], koff), s[1]);
    }

    // online softmax, per output row (r, half)
#pragma unroll
    for (int r = 0; r < 8; ++r) {
      int m = half * 8 + r;
      float w0 = Ws[wave * 16 + m][nc];
      float w1 = Ws[wave * 16 + m][16 + nc];
      float e0 = s[0][r] * scale + __logf(fmaxf(w0, 0.f));
      float e1 = s[1][r] * scale + __logf(fmaxf(w1, 0.f));
      float mloc = fmaxf(e0, e1);
#pragma unroll
      for (int off = 1; off < 16; off <<= 1)
        mloc = fmaxf(mloc, __shfl_xor(mloc, off, 32));
      float mnew = fmaxf(mrun[r], mloc);
      float corr = __expf(mrun[r] - mnew);
      float p0 = __expf(e0 - mnew);
      float p1 = __expf(e1 - mnew);
      float ps = p0 + p1;
#pragma unroll
      for (int off = 1; off < 16; off <<= 1)
        ps += __shfl_xor(ps, off, 32);
      lrun[r] = lrun[r] * corr + ps;
      mrun[r] = mnew;
#pragma unroll
      for (int t = 0; t < 8; ++t) oacc[t][r] *= corr;
      Ps[wave][m][nc]      = f2bf(p0);     // C-layout -> LDS -> A-layout bounce
      Ps[wave][m][16 + nc] = f2bf(p1);
    }

    // O(16x128) += P(16x32) @ V(32x128); B cols come from transposed Vt
    v16bf ap = load_frag(&Ps[wave][nc][0], koff);
#pragma unroll
    for (int t = 0; t < 8; ++t)
      oacc[t] = wmma_bf16(ap, load_frag(&Vt[t * 16 + nc][0], koff), oacc[t]);
    __syncthreads();
  }

  // normalize, add q residual, store f32 merged-head layout
#pragma unroll
  for (int r = 0; r < 8; ++r) {
    float inv = 1.f / lrun[r];
    int row = qrow0 + half * 8 + r;
    const size_t base = ((size_t)b * NQ_ + row) * DV_ + h * DH_;
#pragma unroll
    for (int t = 0; t < 8; ++t) {
      int col = t * 16 + nc;
      O[base + col] = oacc[t][r] * inv + bf2f(qb[base + col]);
    }
  }
}

// ---------------------------------------------------------------------------
// Rowwise LayerNorm over DV=512; optional bf16 copy for the next GEMM.
// ---------------------------------------------------------------------------
__global__ __launch_bounds__(256, 4)
void ln_kernel(const float* __restrict__ X, const float* __restrict__ g,
               const float* __restrict__ bta, float* __restrict__ Y,
               bf16* __restrict__ Ybf) {
  __shared__ float red[16];
  const int tid = threadIdx.x;
  const int wave = tid >> 5, lane = tid & 31;
  const size_t row = blockIdx.x;
  const float* xr = X + row * DV_;
  float x0 = xr[tid], x1 = xr[tid + 256];
  float s = x0 + x1;
#pragma unroll
  for (int off = 1; off < 32; off <<= 1) s += __shfl_xor(s, off, 32);
  if (lane == 0) red[wave] = s;
  __syncthreads();
  float tot = 0.f;
#pragma unroll
  for (int i = 0; i < 8; ++i) tot += red[i];
  float mean = tot * (1.f / DV_);
  float d0 = x0 - mean, d1 = x1 - mean;
  float s2 = d0 * d0 + d1 * d1;
#pragma unroll
  for (int off = 1; off < 32; off <<= 1) s2 += __shfl_xor(s2, off, 32);
  if (lane == 0) red[8 + wave] = s2;
  __syncthreads();
  float tot2 = 0.f;
#pragma unroll
  for (int i = 0; i < 8; ++i) tot2 += red[8 + i];
  float rstd = rsqrtf(tot2 * (1.f / DV_) + 1e-5f);
  float y0 = d0 * rstd * g[tid] + bta[tid];
  float y1 = d1 * rstd * g[tid + 256] + bta[tid + 256];
  Y[row * DV_ + tid] = y0;
  Y[row * DV_ + tid + 256] = y1;
  if (Ybf) {
    Ybf[row * DV_ + tid] = f2bf(y0);
    Ybf[row * DV_ + tid + 256] = f2bf(y1);
  }
}

// ---------------------------------------------------------------------------
// FFN: Y(f32) = Res + relu(Xbf @ Wo^T + bo); same tiling as proj kernel.
// ---------------------------------------------------------------------------
__global__ __launch_bounds__(256, 2)
void ffn_kernel(const bf16* __restrict__ Xbf, const float* __restrict__ W,
                const float* __restrict__ bias, const float* __restrict__ Res,
                float* __restrict__ Y) {
  __shared__ __align__(16) bf16 As[128][32];
  __shared__ __align__(16) bf16 Bs[64][32];
  const int tid  = threadIdx.x;
  const int wave = tid >> 5, lane = tid & 31;
  const int half = lane >> 4, nc = lane & 15;
  const int m0 = blockIdx.x * 128, n0 = blockIdx.y * 64;
  const int koff = half * 8;
  v8f acc[4] = {};
  for (int k0 = 0; k0 < DV_; k0 += 32) {
#pragma unroll
    for (int i = 0; i < 2; ++i) {          // stage A: 128x32 bf16 direct copy
      int idx = tid + i * 256;             // 512 chunks of 8 bf16
      int r = idx >> 2, c8 = idx & 3;
      *(v8bf*)(&As[r][c8 * 8]) =
          *(const v8bf*)(Xbf + (size_t)(m0 + r) * DV_ + k0 + c8 * 8);
    }
#pragma unroll
    for (int i = 0; i < 2; ++i) {          // stage B: 64x32 f32 -> bf16
      int idx = tid + i * 256;
      int r = idx >> 3, c4 = idx & 7;
      float4 f = ((const float4*)(W + (size_t)(n0 + r) * DV_ + k0))[c4];
      bf16* dst = &Bs[r][c4 * 4];
      dst[0] = f2bf(f.x); dst[1] = f2bf(f.y); dst[2] = f2bf(f.z); dst[3] = f2bf(f.w);
    }
    __syncthreads();
    v16bf a = load_frag(&As[wave * 16 + nc][0], koff);
#pragma unroll
    for (int t = 0; t < 4; ++t)
      acc[t] = wmma_bf16(a, load_frag(&Bs[t * 16 + nc][0], koff), acc[t]);
    __syncthreads();
  }
#pragma unroll
  for (int t = 0; t < 4; ++t) {
    int col = n0 + t * 16 + nc;
    float bv = bias[col];
#pragma unroll
    for (int r = 0; r < 8; ++r) {
      int row = m0 + wave * 16 + half * 8 + r;
      size_t idx = (size_t)row * DV_ + col;
      Y[idx] = Res[idx] + fmaxf(acc[t][r] + bv, 0.f);
    }
  }
}

// ---------------------------------------------------------------------------
extern "C" void kernel_launch(void* const* d_in, const int* in_sizes, int n_in,
                              void* d_out, int out_size, void* d_ws, size_t ws_size,
                              hipStream_t stream) {
  const float* Q   = (const float*)d_in[0];
  const float* K   = (const float*)d_in[1];
  const float* wts = (const float*)d_in[2];
  const float* Wq  = (const float*)d_in[3];
  const float* bq  = (const float*)d_in[4];
  const float* Wk  = (const float*)d_in[5];
  const float* bk  = (const float*)d_in[6];
  const float* Wv  = (const float*)d_in[7];
  const float* bv  = (const float*)d_in[8];
  const float* Wo  = (const float*)d_in[9];
  const float* bo  = (const float*)d_in[10];
  const float* g0  = (const float*)d_in[11];
  const float* be0 = (const float*)d_in[12];
  const float* g1  = (const float*)d_in[13];
  const float* be1 = (const float*)d_in[14];
  float* out = (float*)d_out;

  char* ws = (char*)d_ws;
  const size_t NE = (size_t)MROWS * DV_;        // 8.39M elements
  bf16*  qb   = (bf16*)ws;  ws += NE * 2;
  bf16*  kb   = (bf16*)ws;  ws += NE * 2;
  bf16*  vbuf = (bf16*)ws;  ws += NE * 2;
  float* Oat  = (float*)ws; ws += NE * 4;
  float* Xn   = (float*)ws; ws += NE * 4;
  bf16*  Xnbf = (bf16*)ws;  ws += NE * 2;
  float* Z    = (float*)ws; ws += NE * 4;

  dim3 gGemm(MROWS / 128, DV_ / 64);
  proj_f32_kernel<<<gGemm, 256, 0, stream>>>(Q, Wq, bq, qb);
  proj_f32_kernel<<<gGemm, 256, 0, stream>>>(K, Wk, bk, kb);
  proj_f32_kernel<<<gGemm, 256, 0, stream>>>(K, Wv, bv, vbuf);

  dim3 gAttn(NQ_ / 64, H_ * B_SZ);
  attn_kernel<<<gAttn, 128, 0, stream>>>(qb, kb, vbuf, wts, Oat);

  ln_kernel<<<MROWS, 256, 0, stream>>>(Oat, g0, be0, Xn, Xnbf);
  ffn_kernel<<<gGemm, 256, 0, stream>>>(Xnbf, Wo, bo, Xn, Z);
  ln_kernel<<<MROWS, 256, 0, stream>>>(Z, g1, be1, out, (bf16*)nullptr);
}
